// Model_15083925143881
// MI455X (gfx1250) — compile-verified, matrix operand order checked by reading, not run on hardware
//
#include <hip/hip_runtime.h>
#include <hip/hip_bf16.h>

#define BS 16
#define L  512
#define D  1024
#define H  16
#define DK 64

typedef __attribute__((ext_vector_type(16))) __bf16 v16bf;
typedef __attribute__((ext_vector_type(8)))  float  v8f;

// ---------------------------------------------------------------------------
// WMMA fragment helpers (layouts per CDNA5 ISA 7.12.2, wave32)
// A (16x32, bf16): lane 0-15 = row lane, K in {kb..kb+7} u {kb+16..kb+23}, kb=(lane>>4)*8
// B (32x16, bf16): lane = column, same K split => loading B from an N-major
//                  (transposed) source is exactly the A-pattern load.
// C/D (16x16, f32): VGPR r -> M = r + 8*(lane>=16), N = lane&15
// ---------------------------------------------------------------------------

__device__ __forceinline__ v8f wmma_bf16(v16bf a, v16bf b, v8f c) {
    return __builtin_amdgcn_wmma_f32_16x16x32_bf16(
        /*neg_a=*/false, a, /*neg_b=*/false, b,
        /*c_mod=*/(short)0, c, /*reuse_a=*/false, /*reuse_b=*/false);
}

// A-pattern fragment from row-major f32 source (converted to bf16)
__device__ __forceinline__ v16bf frag_f32(const float* __restrict__ p, int ld) {
    const int lane = threadIdx.x & 31;
    const int row  = lane & 15;
    const int kb   = (lane >> 4) << 3;
    const float* r0 = p + (size_t)row * ld + kb;
    v16bf a;
#pragma unroll
    for (int e = 0; e < 8; ++e) a[e] = (__bf16)r0[e];
#pragma unroll
    for (int e = 0; e < 8; ++e) a[8 + e] = (__bf16)r0[16 + e];
    return a;
}

// A-pattern fragment from row-major bf16 source (contiguous 16B x2 per lane)
__device__ __forceinline__ v16bf frag_bf16(const __bf16* __restrict__ p, int ld) {
    const int lane = threadIdx.x & 31;
    const int row  = lane & 15;
    const int kb   = (lane >> 4) << 3;
    const __bf16* r0 = p + (size_t)row * ld + kb;
    v16bf a;
#pragma unroll
    for (int e = 0; e < 8; ++e) a[e] = r0[e];
#pragma unroll
    for (int e = 0; e < 8; ++e) a[8 + e] = r0[16 + e];
    return a;
}

// ---------------------------------------------------------------------------
// Kernel 0: weight transpose + convert:  Wt[n][k] = (bf16) W[k][n]
// LDS-tiled so both global read and write are coalesced. grid (64,64), block (16,16)
// ---------------------------------------------------------------------------
__global__ void __launch_bounds__(256)
wtrans_kernel(const float* __restrict__ W, __bf16* __restrict__ Wt) {
    __shared__ __bf16 t[16][17];
    const int tx = threadIdx.x, ty = threadIdx.y;
    const int k0 = blockIdx.x << 4;
    const int n0 = blockIdx.y << 4;
    t[ty][tx] = (__bf16)W[(size_t)(k0 + ty) * D + n0 + tx];
    __syncthreads();
    Wt[(size_t)(n0 + ty) * D + k0 + tx] = t[tx][ty];
}

// ---------------------------------------------------------------------------
// Kernel 1: projection GEMM  C[8192x1024] = X[8192x1024] * W + bias (W given
// transposed bf16).  Wave tile = 16 rows x 64 cols (one head-aligned column
// block): 4 accumulators reuse one A fragment per k-step.
// Output scattered bf16 with per-head strides:
//   out[ (b*H+h)*L*DK + l*sL + d*sD ]   (q/k: sL=DK,sD=1   v: sL=1,sD=L)
// grid (512, 4), block (32, 4)
// ---------------------------------------------------------------------------
__global__ void __launch_bounds__(128)
proj_kernel(const float* __restrict__ X, const __bf16* __restrict__ Wt,
            const float* __restrict__ bias, __bf16* __restrict__ out,
            int sL, int sD, float scale) {
    const int row0 = blockIdx.x << 4;                                  // 0..8191
    const int colTile = (blockIdx.y * blockDim.y + threadIdx.y);       // 0..15
    const int col0 = colTile << 6;                                     // 64-wide

    v8f a0 = {}, a1 = {}, a2 = {}, a3 = {};
    const float*  xrow = X + (size_t)row0 * D;
    const __bf16* w0 = Wt + (size_t)(col0      ) * D;
    const __bf16* w1 = Wt + (size_t)(col0 + 16 ) * D;
    const __bf16* w2 = Wt + (size_t)(col0 + 32 ) * D;
    const __bf16* w3 = Wt + (size_t)(col0 + 48 ) * D;
#pragma unroll 2
    for (int kk = 0; kk < D; kk += 32) {
        const v16bf a = frag_f32(xrow + kk, D);
        a0 = wmma_bf16(a, frag_bf16(w0 + kk, D), a0);
        a1 = wmma_bf16(a, frag_bf16(w1 + kk, D), a1);
        a2 = wmma_bf16(a, frag_bf16(w2 + kk, D), a2);
        a3 = wmma_bf16(a, frag_bf16(w3 + kk, D), a3);
    }

    const int lane = threadIdx.x & 31;
    const int n  = lane & 15;
    const int mo = (lane >> 4) << 3;
    const int h  = col0 >> 6;                       // whole tile in one head
    const v8f* accs[4] = { &a0, &a1, &a2, &a3 };
#pragma unroll
    for (int jj = 0; jj < 4; ++jj) {
        const int dd = (jj << 4) + n;               // dim inside head
        const float bj = bias[col0 + dd];
#pragma unroll
        for (int r = 0; r < 8; ++r) {
            const int m  = row0 + mo + r;           // flattened (b, l)
            const int b_ = m >> 9;
            const int l  = m & (L - 1);
            const size_t headBase = ((size_t)b_ * H + h) * (size_t)L * DK;
            const float v = ((*accs[jj])[r] + bj) * scale;
            out[headBase + (size_t)l * sL + (size_t)dd * sD] = (__bf16)v;
        }
    }
}

// ---------------------------------------------------------------------------
// Kernel 2: per-(b,h) attention. One wave per 16-query tile.
//   scores = (q k^T)*matrix (scale pre-folded into q), written f32 to d_out
//   softmax over LDS 16x512 tile, then attn(bf16) x V^T -> bf16 context buf.
// V is stored transposed [B,H,DK,L] so its B fragments are contiguous.
// grid (32, 256), block (32)
// ---------------------------------------------------------------------------
__global__ void __launch_bounds__(32)
attn_kernel(const __bf16* __restrict__ qb, const __bf16* __restrict__ kbuf,
            const __bf16* __restrict__ vbt, const float* __restrict__ matrix,
            float* __restrict__ scores_out, __bf16* __restrict__ ctx) {
    __shared__ float  sc[16][L];   // 32 KB f32 scores tile
    __shared__ __bf16 at[16][L];   // 16 KB bf16 attn weights

    const int bh = blockIdx.y;     // 0..255
    const int b  = bh >> 4;
    const int h  = bh & 15;
    const int l0 = blockIdx.x << 4;

    const size_t headOff = ((size_t)b * H + h) * L;
    const __bf16* q0 = qb   + (headOff + l0) * DK;
    const __bf16* k0 = kbuf + headOff * DK;
    const __bf16* v0 = vbt  + headOff * DK;            // [DK][L] per head
    const float*  m0 = matrix     + (headOff + l0) * L;
    float*        s0 = scores_out + (headOff + l0) * L;

    const int lane = threadIdx.x;
    const int n  = lane & 15;
    const int mo = (lane >> 4) << 3;

    // q tile fragments (K=64 -> two 16x32 A fragments), reused for all key tiles
    const v16bf qa0 = frag_bf16(q0, DK);
    const v16bf qa1 = frag_bf16(q0 + 32, DK);

    for (int ct = 0; ct < L / 16; ++ct) {
        const __bf16* kt = k0 + (size_t)(ct << 4) * DK;
        // prefetch the next streamed key tile and matrix block (global_prefetch)
        if (ct + 1 < L / 16) {
            __builtin_prefetch(kt + (size_t)16 * DK, 0, 0);
            __builtin_prefetch(&m0[(size_t)mo * L + ((ct + 1) << 4)], 0, 0);
        }
        v8f acc = {};
        // B = k^T: per-lane gather of key rows == A-pattern load on K matrix
        acc = wmma_bf16(qa0, frag_bf16(kt, DK), acc);
        acc = wmma_bf16(qa1, frag_bf16(kt + 32, DK), acc);
#pragma unroll
        for (int r = 0; r < 8; ++r) {
            const size_t idx = (size_t)(mo + r) * L + (ct << 4) + n;
            const float s = acc[r] * m0[idx];
            s0[idx] = s;                           // scores output
            sc[mo + r][(ct << 4) + n] = s;
        }
    }
    __syncthreads();

    // softmax: lanes 0..15 each own one query row
    if (lane < 16) {
        float mx = -3.402823466e38f;
        for (int j = 0; j < L; ++j) mx = fmaxf(mx, sc[lane][j]);
        float sum = 0.f;
        for (int j = 0; j < L; ++j) {
            const float e = __expf(sc[lane][j] - mx);
            sum += e;
            sc[lane][j] = e;
        }
        const float inv = 1.f / sum;
        for (int j = 0; j < L; ++j) at[lane][j] = (__bf16)(sc[lane][j] * inv);
    }
    __syncthreads();

    // out[16 x 64] = attn[16 x 512] x V[512 x 64]; B frags from V^T contiguous
    v8f o0 = {}, o1 = {}, o2 = {}, o3 = {};
    for (int kk = 0; kk < L; kk += 32) {
        const v16bf a = frag_bf16(&at[0][0] + kk, L);
        o0 = wmma_bf16(a, frag_bf16(v0 + (size_t) 0 * L + kk, L), o0);
        o1 = wmma_bf16(a, frag_bf16(v0 + (size_t)16 * L + kk, L), o1);
        o2 = wmma_bf16(a, frag_bf16(v0 + (size_t)32 * L + kk, L), o2);
        o3 = wmma_bf16(a, frag_bf16(v0 + (size_t)48 * L + kk, L), o3);
    }

    // scatter to [B*L, D] bf16 context buffer at columns h*64 .. h*64+63
    __bf16* od = ctx + ((size_t)b * L + l0) * D + (size_t)h * DK;
#pragma unroll
    for (int r = 0; r < 8; ++r) {
        const size_t ro = (size_t)(mo + r) * D;
        od[ro +  0 + n] = (__bf16)o0[r];
        od[ro + 16 + n] = (__bf16)o1[r];
        od[ro + 32 + n] = (__bf16)o2[r];
        od[ro + 48 + n] = (__bf16)o3[r];
    }
}

// ---------------------------------------------------------------------------
// Kernel 3: output projection  out[8192x1024] = ctx(bf16) * W_O + b_O (f32)
// W_O given transposed bf16. Wave tile 16x64. grid (512, 4), block (32, 4)
// ---------------------------------------------------------------------------
__global__ void __launch_bounds__(128)
oproj_kernel(const __bf16* __restrict__ Xc, const __bf16* __restrict__ Wt,
             const float* __restrict__ bias, float* __restrict__ out) {
    const int row0 = blockIdx.x << 4;
    const int col0 = (blockIdx.y * blockDim.y + threadIdx.y) << 6;

    v8f a0 = {}, a1 = {}, a2 = {}, a3 = {};
    const __bf16* xrow = Xc + (size_t)row0 * D;
    const __bf16* w0 = Wt + (size_t)(col0      ) * D;
    const __bf16* w1 = Wt + (size_t)(col0 + 16 ) * D;
    const __bf16* w2 = Wt + (size_t)(col0 + 32 ) * D;
    const __bf16* w3 = Wt + (size_t)(col0 + 48 ) * D;
#pragma unroll 2
    for (int kk = 0; kk < D; kk += 32) {
        const v16bf a = frag_bf16(xrow + kk, D);
        a0 = wmma_bf16(a, frag_bf16(w0 + kk, D), a0);
        a1 = wmma_bf16(a, frag_bf16(w1 + kk, D), a1);
        a2 = wmma_bf16(a, frag_bf16(w2 + kk, D), a2);
        a3 = wmma_bf16(a, frag_bf16(w3 + kk, D), a3);
    }

    const int lane = threadIdx.x & 31;
    const int n  = lane & 15;
    const int mo = (lane >> 4) << 3;
    const v8f* accs[4] = { &a0, &a1, &a2, &a3 };
#pragma unroll
    for (int jj = 0; jj < 4; ++jj) {
        const int j = col0 + (jj << 4) + n;
        const float bj = bias[j];
#pragma unroll
        for (int r = 0; r < 8; ++r) {
            const int m = row0 + mo + r;
            out[(size_t)m * D + j] = (*accs[jj])[r] + bj;
        }
    }
}

// ---------------------------------------------------------------------------
extern "C" void kernel_launch(void* const* d_in, const int* in_sizes, int n_in,
                              void* d_out, int out_size, void* d_ws, size_t ws_size,
                              hipStream_t stream) {
    const float* Q      = (const float*)d_in[0];
    const float* W_Q    = (const float*)d_in[1];
    const float* b_Q    = (const float*)d_in[2];
    const float* W_K    = (const float*)d_in[3];
    const float* b_K    = (const float*)d_in[4];
    const float* W_V    = (const float*)d_in[5];
    const float* b_V    = (const float*)d_in[6];
    const float* matrix = (const float*)d_in[7];
    const float* W_O    = (const float*)d_in[8];
    const float* b_O    = (const float*)d_in[9];

    float* out    = (float*)d_out;                 // [BS, L, D]
    float* scores = out + (size_t)BS * L * D;      // [BS, H, L, L]

    const size_t NE = (size_t)BS * L * D;          // 8,388,608 elements
    const size_t NW = (size_t)D * D;               // 1,048,576 elements
    __bf16* qb  = (__bf16*)d_ws;                   // [B,H,L,DK]  16 MB
    __bf16* kb  = qb  + NE;                        // [B,H,L,DK]  16 MB
    __bf16* vbt = kb  + NE;                        // [B,H,DK,L]  16 MB
    __bf16* ctx = vbt + NE;                        // [B*L, D]    16 MB
    __bf16* WQt = ctx + NE;                        // 4x [N,K] bf16, 2 MB each
    __bf16* WKt = WQt + NW;
    __bf16* WVt = WKt + NW;
    __bf16* WOt = WVt + NW;

    // --- transpose + bf16-convert the four weight matrices ---
    const dim3 tGrid(D / 16, D / 16), tBlk(16, 16);
    hipLaunchKernelGGL(wtrans_kernel, tGrid, tBlk, 0, stream, W_Q, WQt);
    hipLaunchKernelGGL(wtrans_kernel, tGrid, tBlk, 0, stream, W_K, WKt);
    hipLaunchKernelGGL(wtrans_kernel, tGrid, tBlk, 0, stream, W_V, WVt);
    hipLaunchKernelGGL(wtrans_kernel, tGrid, tBlk, 0, stream, W_O, WOt);

    // --- projections (scale = DK^-0.5 = 0.125 folded into Q) ---
    const dim3 gBlk(32, 4);
    const dim3 gGrid(BS * L / 16, D / 64 / 4);
    hipLaunchKernelGGL(proj_kernel, gGrid, gBlk, 0, stream, Q, WQt, b_Q, qb,
                       DK, 1, 0.125f);
    hipLaunchKernelGGL(proj_kernel, gGrid, gBlk, 0, stream, Q, WKt, b_K, kb,
                       DK, 1, 1.0f);
    hipLaunchKernelGGL(proj_kernel, gGrid, gBlk, 0, stream, Q, WVt, b_V, vbt,
                       1, L, 1.0f);   // V stored transposed per head

    // --- attention (scores + softmax + attn x V) ---
    hipLaunchKernelGGL(attn_kernel, dim3(L / 16, BS * H), dim3(32), 0, stream,
                       qb, kb, vbt, matrix, scores, ctx);

    // --- output projection ---
    hipLaunchKernelGGL(oproj_kernel, gGrid, gBlk, 0, stream, ctx, WOt, b_O, out);
}
